// CRISPRO_Apex_4827543241082
// MI455X (gfx1250) — matrix operand order repaired and checked by reading, not compiled, not run on hardware
//
#include <hip/hip_runtime.h>
#include <math.h>

#define D_MODELC   512
#define N_LAYERSC  6
#define D_STATEC   128
#define HEADDIMC   64
#define D_INNERC   1024
#define NHEADSC    16
#define CONV_DIMC  1280
#define D_IN_PROJC 2320
#define BATCHC     2
#define SEQC       4096
#define TOKENSC    (BATCHC*SEQC)
#define DKWC       4
#define CHUNKC     256
#define NCHUNKC    (SEQC/CHUNKC)

typedef __attribute__((ext_vector_type(16))) __bf16 v16bf;
typedef __attribute__((ext_vector_type(8)))  float  v8f;
typedef unsigned short u16;

union Frag { uint4 q[2]; v16bf v; };

// ---------- gfx1250 async-to-LDS (guarded; falls back to plain load+store) ----------
#if defined(__has_builtin)
#  if __has_builtin(__builtin_amdgcn_global_load_async_to_lds_b32)
#    define ASYNC_LDS 1
#  endif
#endif

typedef __attribute__((address_space(1))) int as1_int;   // "__device__ int *" in clang diagnostics
typedef __attribute__((address_space(3))) int as3_int;   // LDS int *

__device__ __forceinline__ void stage_lds_b32(const float* gsrc, float* ldst){
#ifdef ASYNC_LDS
  __builtin_amdgcn_global_load_async_to_lds_b32((as1_int*)gsrc, (as3_int*)ldst, 0, 0);
#else
  *ldst = *gsrc;
#endif
}
__device__ __forceinline__ void async_wait0(){
#ifdef ASYNC_LDS
#  if __has_builtin(__builtin_amdgcn_s_wait_asynccnt)
  __builtin_amdgcn_s_wait_asynccnt(0);
#  else
  asm volatile("s_wait_asynccnt 0x0" ::: "memory");
#  endif
#endif
}

// ---------- helpers ----------
__device__ __forceinline__ __bf16 bfc(float f){ return (__bf16)f; }
__device__ __forceinline__ u16 bfbits(float f){
  union { __bf16 b; u16 s; } u; u.b = (__bf16)f; return u.s;
}
// A-fragment K offset for element i (16-bit A 16x32 layout, ISA 7.12.2)
__device__ __forceinline__ int aofs(int i, int half){ return (i < 8) ? (half*8 + i) : (8 + half*8 + i); }
__device__ __forceinline__ float sigf(float x){ return 1.f/(1.f+__expf(-x)); }
__device__ __forceinline__ float siluf(float x){ return x/(1.f+__expf(-x)); }

template<int ACT> __device__ __forceinline__ float actf(float x){
  if (ACT == 1) return sigf(x);                                  // sigmoid
  if (ACT == 2) return 0.5f*x*(1.f + erff(x*0.70710678118f));    // exact gelu
  return x;
}

// ---------- bf16 packing ----------
__global__ void k_pack(const float* __restrict__ a, u16* __restrict__ o, long n){
  long i = (long)blockIdx.x*256 + threadIdx.x;
  if (i < n) o[i] = bfbits(a[i]);
}

// o[n*K + k] = bf16(w[k*N + n])  (transpose via LDS tile for coalescing)
__global__ void k_packT(const float* __restrict__ w, u16* __restrict__ o, int K, int N){
  __shared__ float tile[32][33];
  int kb = blockIdx.y*32, nb = blockIdx.x*32;
  int tx = threadIdx.x & 31, ty = threadIdx.x >> 5;   // ty: 0..7
  for (int r = ty; r < 32; r += 8){
    int k = kb + r, n = nb + tx;
    tile[r][tx] = (k < K && n < N) ? w[(long)k*N + n] : 0.f;
  }
  __syncthreads();
  for (int r = ty; r < 32; r += 8){
    int n = nb + r, k = kb + tx;
    if (n < N && k < K) o[(long)n*K + k] = bfbits(tile[tx][r]);
  }
}

// ---------- bf16 WMMA GEMM: C = act(A[M,K] @ W[K,N] + bias), W packed transposed [N,K] ----------
template<int ACT>
__global__ __launch_bounds__(256) void gemm_bf16(
    const u16* __restrict__ A,    // [M,K] bf16 row-major
    const u16* __restrict__ Bt,   // [N,K] bf16 (pre-transposed)
    const float* __restrict__ bias,
    float* __restrict__ C, int ldc,
    int M, int N, int K)
{
  int lane = threadIdx.x & 31, wave = threadIdx.x >> 5;
  int wm = wave >> 1, wn = wave & 1;
  int m0 = blockIdx.y*64 + wm*16;
  int n0 = blockIdx.x*64 + wn*32;
  int half = lane >> 4, l16 = lane & 15;
  int mrow = m0 + l16; if (mrow > M-1) mrow = M-1;       // clamp (stores guarded)
  int c0 = n0 + l16, c1 = c0 + 16;
  int c0c = (c0 > N-1) ? N-1 : c0;
  int c1c = (c1 > N-1) ? N-1 : c1;
  const u16* Ap  = A  + (long)mrow*K + half*16;
  const u16* B0p = Bt + (long)c0c*K + half*16;
  const u16* B1p = Bt + (long)c1c*K + half*16;
  v8f acc0 = {}; v8f acc1 = {};
  for (int k0 = 0; k0 < K; k0 += 32){
    __builtin_prefetch(Ap  + k0 + 128, 0, 0);
    __builtin_prefetch(B0p + k0 + 128, 0, 0);
    Frag fa, fb0, fb1;
    fa.q[0]  = *(const uint4*)(Ap  + k0);
    fa.q[1]  = *(const uint4*)(Ap  + k0 + 8);
    fb0.q[0] = *(const uint4*)(B0p + k0);
    fb0.q[1] = *(const uint4*)(B0p + k0 + 8);
    fb1.q[0] = *(const uint4*)(B1p + k0);
    fb1.q[1] = *(const uint4*)(B1p + k0 + 8);
    acc0 = __builtin_amdgcn_wmma_f32_16x16x32_bf16(false, fa.v, false, fb0.v, (short)0, acc0, false, false);
    acc1 = __builtin_amdgcn_wmma_f32_16x16x32_bf16(false, fa.v, false, fb1.v, (short)0, acc1, false, false);
  }
  #pragma unroll
  for (int r = 0; r < 8; ++r){
    int row = m0 + half*8 + r;
    if (row >= M) continue;
    if (c0 < N) C[(long)row*ldc + c0] = actf<ACT>(acc0[r] + (bias ? bias[c0] : 0.f));
    if (c1 < N) C[(long)row*ldc + c1] = actf<ACT>(acc1[r] + (bias ? bias[c1] : 0.f));
  }
}

// ---------- embedding ----------
__global__ void k_embed(const int* __restrict__ seq, const float* __restrict__ emb, float* __restrict__ x){
  long t = blockIdx.x; int v = seq[t];
  for (int i = threadIdx.x; i < D_MODELC; i += 256) x[t*D_MODELC + i] = emb[(long)v*D_MODELC + i];
}

// ---------- dilated epi conv ----------
__global__ void k_epiconv(const float* __restrict__ epi, const float* __restrict__ w,
                          const float* __restrict__ bv, float* __restrict__ ef){
  int t = blockIdx.x; int b = t / SEQC; int tl = t - b*SEQC;
  for (int c = threadIdx.x; c < D_MODELC; c += 256){
    int g = c >> 7, oc = c & 127;
    int dil = 1 << g;
    float acc = bv[g*128 + oc];
    #pragma unroll
    for (int tr = 0; tr < 5; ++tr){
      #pragma unroll
      for (int k = 0; k < 3; ++k){
        int ts = tl + (k - 1)*dil;
        if (ts >= 0 && ts < SEQC)
          acc += w[((g*128 + oc)*5 + tr)*3 + k] * epi[((long)(b*SEQC + ts))*5 + tr];
      }
    }
    ef[(long)t*D_MODELC + c] = acc;
  }
}

// ---------- LN / RMS kernels ----------
__global__ void k_fuse_ln(float* __restrict__ x, const float* __restrict__ gate,
                          const float* __restrict__ fp, const float* __restrict__ g,
                          const float* __restrict__ b){
  __shared__ float red[256];
  long t = blockIdx.x; int tid = threadIdx.x;
  float v0 = x[t*512 + tid]       + gate[t*512 + tid]       * fp[t*512 + tid];
  float v1 = x[t*512 + 256 + tid] + gate[t*512 + 256 + tid] * fp[t*512 + 256 + tid];
  red[tid] = v0 + v1; __syncthreads();
  for (int s = 128; s > 0; s >>= 1){ if (tid < s) red[tid] += red[tid + s]; __syncthreads(); }
  float mu = red[0] * (1.f/512.f); __syncthreads();
  float d0 = v0 - mu, d1 = v1 - mu;
  red[tid] = d0*d0 + d1*d1; __syncthreads();
  for (int s = 128; s > 0; s >>= 1){ if (tid < s) red[tid] += red[tid + s]; __syncthreads(); }
  float inv = rsqrtf(red[0] * (1.f/512.f) + 1e-5f);
  x[t*512 + tid]       = d0*inv*g[tid]       + b[tid];
  x[t*512 + 256 + tid] = d1*inv*g[256 + tid] + b[256 + tid];
}

__global__ void k_ln_out(const float* __restrict__ x, const float* __restrict__ g,
                         const float* __restrict__ b, float* __restrict__ o){
  __shared__ float red[256];
  long t = blockIdx.x; int tid = threadIdx.x;
  float v0 = x[t*512 + tid], v1 = x[t*512 + 256 + tid];
  red[tid] = v0 + v1; __syncthreads();
  for (int s = 128; s > 0; s >>= 1){ if (tid < s) red[tid] += red[tid + s]; __syncthreads(); }
  float mu = red[0] * (1.f/512.f); __syncthreads();
  float d0 = v0 - mu, d1 = v1 - mu;
  red[tid] = d0*d0 + d1*d1; __syncthreads();
  for (int s = 128; s > 0; s >>= 1){ if (tid < s) red[tid] += red[tid + s]; __syncthreads(); }
  float inv = rsqrtf(red[0] * (1.f/512.f) + 1e-5f);
  o[t*512 + tid]       = d0*inv*g[tid]       + b[tid];
  o[t*512 + 256 + tid] = d1*inv*g[256 + tid] + b[256 + tid];
}

__global__ void k_rms(const float* __restrict__ x, const float* __restrict__ g, float* __restrict__ o){
  __shared__ float red[256];
  long t = blockIdx.x; int tid = threadIdx.x;
  float a = x[t*512 + tid], b = x[t*512 + 256 + tid];
  red[tid] = a*a + b*b; __syncthreads();
  for (int s = 128; s > 0; s >>= 1){ if (tid < s) red[tid] += red[tid + s]; __syncthreads(); }
  float inv = rsqrtf(red[0] * (1.f/512.f) + 1e-5f);
  o[t*512 + tid]       = a*inv*g[tid];
  o[t*512 + 256 + tid] = b*inv*g[256 + tid];
}

// ---------- flips / combines ----------
__global__ void k_flip(const float* __restrict__ in, float* __restrict__ out){
  int t = blockIdx.x; int b = t / SEQC; int tl = t - b*SEQC;
  long src = (long)(b*SEQC + (SEQC - 1 - tl)) * D_MODELC;
  for (int i = threadIdx.x; i < D_MODELC; i += 256) out[(long)t*D_MODELC + i] = in[src + i];
}

__global__ void k_combine(const float* __restrict__ x, const float* __restrict__ f,
                          const float* __restrict__ bw, float* __restrict__ xm){
  int t = blockIdx.x; int b = t / SEQC; int tl = t - b*SEQC;
  long fb = (long)(b*SEQC + (SEQC - 1 - tl)) * D_MODELC;
  for (int i = threadIdx.x; i < D_MODELC; i += 256)
    xm[(long)t*D_MODELC + i] = x[(long)t*D_MODELC + i] + f[(long)t*D_MODELC + i] + bw[fb + i];
}

__global__ void k_tanhres(const float* __restrict__ xm, const float* __restrict__ mt, float* __restrict__ x){
  long t = blockIdx.x;
  for (int i = threadIdx.x; i < D_MODELC; i += 256)
    x[t*D_MODELC + i] = xm[t*D_MODELC + i] + 0.1f*tanhf(mt[t*D_MODELC + i]);
}

// ---------- causal depthwise conv (K=4) + silu ----------
__global__ void k_conv(const float* __restrict__ zx, const float* __restrict__ cw,
                       const float* __restrict__ cb, float* __restrict__ xbc){
  int t = blockIdx.x; int b = t / SEQC; int tl = t - b*SEQC;
  for (int c = threadIdx.x; c < CONV_DIMC; c += 256){
    float acc = cb[c];
    #pragma unroll
    for (int k = 0; k < DKWC; ++k){
      int ts = tl - (DKWC - 1) + k;
      if (ts >= 0)
        acc += cw[c*DKWC + k] * zx[((long)(b*SEQC + ts))*D_IN_PROJC + D_INNERC + c];
    }
    xbc[(long)t*CONV_DIMC + c] = siluf(acc);
  }
}

// ---------- dt softplus + within-chunk cumulative log-decay ----------
__global__ void k_dtseg(const float* __restrict__ zx, const float* __restrict__ dtb,
                        const float* __restrict__ alog, float* __restrict__ dt, float* __restrict__ seg){
  int bc = blockIdx.x; int h = threadIdx.x;
  if (h >= NHEADSC) return;
  int tb = bc * CHUNKC;
  float Ah = -__expf(alog[h]);
  float db = dtb[h];
  float run = 0.f;
  for (int q = 0; q < CHUNKC; ++q){
    long tg = tb + q;
    float v = zx[tg*D_IN_PROJC + (D_INNERC + CONV_DIMC) + h] + db;
    float dtv = (v > 20.f) ? v : log1pf(__expf(v));
    dt[tg*NHEADSC + h] = dtv;
    run += dtv * Ah;
    seg[tg*NHEADSC + h] = run;
  }
}

__global__ void k_dtx(const float* __restrict__ dt, const float* __restrict__ xbc, float* __restrict__ dtx){
  long t = blockIdx.x;
  for (int i = threadIdx.x; i < D_INNERC; i += 256)
    dtx[t*D_INNERC + i] = dt[t*NHEADSC + (i >> 6)] * xbc[t*CONV_DIMC + i];
}

// ---------- CB = Cm @ Bm^T per chunk (lower-triangle tiles only) ----------
__global__ __launch_bounds__(256) void k_cb(const float* __restrict__ xbc, float* __restrict__ CB){
  int bc = blockIdx.y;
  int wave = threadIdx.x >> 5, lane = threadIdx.x & 31;
  int tile = blockIdx.x*8 + wave;          // 256 tiles: 16 t-tiles x 16 s-tiles
  int tt = tile >> 4, st = tile & 15;
  if (st > tt) return;                     // causal: upper triangle never read
  int t0 = tt*16, s0c = st*16;
  int tb = bc * CHUNKC;
  int half = lane >> 4, l16 = lane & 15;
  const float* arow = xbc + (long)(tb + t0  + l16)*CONV_DIMC + (D_INNERC + D_STATEC); // Cm row
  const float* brow = xbc + (long)(tb + s0c + l16)*CONV_DIMC + D_INNERC;              // Bm row
  v8f acc = {};
  for (int k0 = 0; k0 < D_STATEC; k0 += 32){
    v16bf af, bfb;
    #pragma unroll
    for (int i = 0; i < 16; ++i) af[i]  = bfc(arow[k0 + half*16 + i]);
    #pragma unroll
    for (int i = 0; i < 16; ++i) bfb[i] = bfc(brow[k0 + half*16 + i]);
    acc = __builtin_amdgcn_wmma_f32_16x16x32_bf16(false, af, false, bfb, (short)0, acc, false, false);
  }
  float* out = CB + (long)bc*CHUNKC*CHUNKC;
  #pragma unroll
  for (int r = 0; r < 8; ++r)
    out[(long)(t0 + half*8 + r)*CHUNKC + s0c + l16] = acc[r];
}

// ---------- SSD intra-chunk: Y_intra = (CB .* Lmat) @ (dt*x) ----------
// seg column for this (chunk, head) staged into LDS once per block (async-to-LDS gather).
__global__ __launch_bounds__(256) void k_yintra(const float* __restrict__ CB, const float* __restrict__ seg,
                                                const float* __restrict__ dtx, float* __restrict__ y){
  int h = blockIdx.z, bc = blockIdx.y;
  int tb = bc * CHUNKC;
  __shared__ float sseg[CHUNKC];
  stage_lds_b32(seg + (long)(tb + threadIdx.x)*NHEADSC + h, &sseg[threadIdx.x]);
  async_wait0();
  __syncthreads();

  int wave = threadIdx.x >> 5, lane = threadIdx.x & 31;
  int tile = blockIdx.x*8 + wave;       // 64 tiles: 16 t-tiles x 4 p-tiles
  int tt = tile >> 2, pt = tile & 3;
  int t0 = tt*16, p0 = pt*16;
  int half = lane >> 4, l16 = lane & 15;
  const float* cbc = CB + (long)bc*CHUNKC*CHUNKC + (long)(t0 + l16)*CHUNKC;
  int tm = t0 + l16;
  float seg_t = sseg[tm];
  v8f acc = {};
  for (int s0 = 0; s0 < t0 + 16; s0 += 32){
    v16bf af, bfb;
    #pragma unroll
    for (int i = 0; i < 16; ++i){
      int s = s0 + half*16 + i;
      float v = (s <= tm) ? cbc[s] * __expf(seg_t - sseg[s]) : 0.f;
      af[i] = bfc(v);
    }
    #pragma unroll
    for (int i = 0; i < 16; ++i){
      int s = s0 + half*16 + i;
      bfb[i] = bfc(dtx[(long)(tb + s)*D_INNERC + h*HEADDIMC + p0 + l16]);
    }
    acc = __builtin_amdgcn_wmma_f32_16x16x32_bf16(false, af, false, bfb, (short)0, acc, false, false);
  }
  #pragma unroll
  for (int r = 0; r < 8; ++r){
    int t = t0 + half*8 + r;
    y[(long)(tb + t)*D_INNERC + h*HEADDIMC + p0 + l16] = acc[r];
  }
}

// ---------- SSD chunk states: states[p,n] = sum_q dend[q]*dtx[q,p]*Bm[q,n] ----------
__global__ __launch_bounds__(256) void k_states(const float* __restrict__ seg, const float* __restrict__ dtx,
                                                const float* __restrict__ xbc, float* __restrict__ st){
  int h = blockIdx.z, bc = blockIdx.y;
  int tb = bc * CHUNKC;
  __shared__ float sseg[CHUNKC];
  stage_lds_b32(seg + (long)(tb + threadIdx.x)*NHEADSC + h, &sseg[threadIdx.x]);
  async_wait0();
  __syncthreads();

  int wave = threadIdx.x >> 5, lane = threadIdx.x & 31;
  int tile = blockIdx.x*8 + wave;       // 32 tiles: 4 p-tiles x 8 n-tiles
  int pt = tile >> 3, nt = tile & 7;
  int p0 = pt*16, n0 = nt*16;
  int half = lane >> 4, l16 = lane & 15;
  float segend = sseg[CHUNKC - 1];
  v8f acc = {};
  for (int k0 = 0; k0 < CHUNKC; k0 += 32){
    v16bf af, bfb;
    #pragma unroll
    for (int i = 0; i < 16; ++i){
      int q = k0 + half*16 + i;
      float dend = __expf(segend - sseg[q]);
      af[i] = bfc(dend * dtx[(long)(tb + q)*D_INNERC + h*HEADDIMC + p0 + l16]);
    }
    #pragma unroll
    for (int i = 0; i < 16; ++i){
      int q = k0 + half*16 + i;
      bfb[i] = bfc(xbc[(long)(tb + q)*CONV_DIMC + D_INNERC + n0 + l16]);
    }
    acc = __builtin_amdgcn_wmma_f32_16x16x32_bf16(false, af, false, bfb, (short)0, acc, false, false);
  }
  long base = ((long)bc*NHEADSC + h)*(HEADDIMC*D_STATEC);
  #pragma unroll
  for (int r = 0; r < 8; ++r){
    int p = p0 + half*8 + r;
    st[base + (long)p*D_STATEC + n0 + l16] = acc[r];
  }
}

// ---------- inter-chunk sequential scan (in place: states -> Hprev) ----------
__global__ void k_scan(const float* __restrict__ seg, float* __restrict__ st){
  int b = blockIdx.x / NHEADSC, h = blockIdx.x % NHEADSC;
  float hreg[32];
  #pragma unroll
  for (int j = 0; j < 32; ++j) hreg[j] = 0.f;
  for (int c = 0; c < NCHUNKC; ++c){
    int bc = b*NCHUNKC + c;
    long base = ((long)bc*NHEADSC + h)*(HEADDIMC*D_STATEC);
    int tb = bc * CHUNKC;
    float cd = __expf(seg[(long)(tb + CHUNKC - 1)*NHEADSC + h]);
    #pragma unroll
    for (int j = 0; j < 32; ++j){
      long idx = base + threadIdx.x + j*256;
      float s = st[idx];
      st[idx] = hreg[j];
      hreg[j] = cd*hreg[j] + s;
    }
  }
}

// ---------- Y_inter = exp(seg)*(Cm @ Hprev^T) + D*xh, accumulated onto Y ----------
__global__ __launch_bounds__(256) void k_yinter(const float* __restrict__ seg, const float* __restrict__ xbc,
                                                const float* __restrict__ st, const float* __restrict__ Dp,
                                                float* __restrict__ y){
  int h = blockIdx.z, bc = blockIdx.y;
  int tb = bc * CHUNKC;
  __shared__ float sseg[CHUNKC];
  stage_lds_b32(seg + (long)(tb + threadIdx.x)*NHEADSC + h, &sseg[threadIdx.x]);
  async_wait0();
  __syncthreads();

  int wave = threadIdx.x >> 5, lane = threadIdx.x & 31;
  int tile = blockIdx.x*8 + wave;       // 64 tiles: 16 t-tiles x 4 p-tiles
  int tt = tile >> 2, pt = tile & 3;
  int t0 = tt*16, p0 = pt*16;
  int half = lane >> 4, l16 = lane & 15;
  const float* arow = xbc + (long)(tb + t0 + l16)*CONV_DIMC + (D_INNERC + D_STATEC);
  const float* brow = st + ((long)bc*NHEADSC + h)*(HEADDIMC*D_STATEC) + (long)(p0 + l16)*D_STATEC;
  v8f acc = {};
  for (int k0 = 0; k0 < D_STATEC; k0 += 32){
    v16bf af, bfb;
    #pragma unroll
    for (int i = 0; i < 16; ++i) af[i]  = bfc(arow[k0 + half*16 + i]);
    #pragma unroll
    for (int i = 0; i < 16; ++i) bfb[i] = bfc(brow[k0 + half*16 + i]);
    acc = __builtin_amdgcn_wmma_f32_16x16x32_bf16(false, af, false, bfb, (short)0, acc, false, false);
  }
  float dph = Dp[h];
  #pragma unroll
  for (int r = 0; r < 8; ++r){
    int t = t0 + half*8 + r;
    long tg = tb + t;
    float sc = __expf(sseg[t]);
    long yi = tg*D_INNERC + h*HEADDIMC + p0 + l16;
    y[yi] = y[yi] + sc*acc[r] + dph * xbc[tg*CONV_DIMC + h*HEADDIMC + p0 + l16];
  }
}

// ---------- y *= silu(z) ----------
__global__ void k_gatemul(const float* __restrict__ zx, float* __restrict__ y){
  long t = blockIdx.x;
  for (int i = threadIdx.x; i < D_INNERC; i += 256)
    y[t*D_INNERC + i] *= siluf(zx[t*D_IN_PROJC + i]);
}

// ---------- mean pool ----------
__global__ void k_pool(const float* __restrict__ xf, float* __restrict__ lat){
  __shared__ float red[256];
  int b = blockIdx.x >> 9, d = blockIdx.x & 511;
  float s = 0.f;
  for (int t = threadIdx.x; t < SEQC; t += 256)
    s += xf[((long)(b*SEQC + t))*D_MODELC + d];
  red[threadIdx.x] = s; __syncthreads();
  for (int st = 128; st > 0; st >>= 1){ if (threadIdx.x < st) red[threadIdx.x] += red[threadIdx.x + st]; __syncthreads(); }
  if (threadIdx.x == 0) lat[b*D_MODELC + d] = red[0] * (1.f/(float)SEQC);
}

// ---------- final 256->1 head ----------
__global__ void k_head2(const float* __restrict__ h1, const float* __restrict__ w2,
                        const float* __restrict__ b2, float* __restrict__ dst){
  __shared__ float red[256];
  int b = blockIdx.x;
  red[threadIdx.x] = h1[b*256 + threadIdx.x] * w2[threadIdx.x]; __syncthreads();
  for (int s = 128; s > 0; s >>= 1){ if (threadIdx.x < s) red[threadIdx.x] += red[threadIdx.x + s]; __syncthreads(); }
  if (threadIdx.x == 0) dst[b] = sigf(red[0] + b2[0]);
}

__global__ void k_copy(const float* __restrict__ s, float* __restrict__ d, int n){
  int i = blockIdx.x*256 + threadIdx.x; if (i < n) d[i] = s[i];
}

// =====================================================================
extern "C" void kernel_launch(void* const* d_in, const int* in_sizes, int n_in,
                              void* d_out, int out_size, void* d_ws, size_t ws_size,
                              hipStream_t stream){
  (void)in_sizes; (void)n_in; (void)out_size; (void)ws_size;
  const int*   dna    = (const int*)  d_in[0];
  const float* epi    = (const float*)d_in[1];
  const float* emb    = (const float*)d_in[2];
  const float* fcw    = (const float*)d_in[3];
  const float* fcb    = (const float*)d_in[4];
  const float* gw     = (const float*)d_in[5];
  const float* gb     = (const float*)d_in[6];
  const float* fw     = (const float*)d_in[7];
  const float* fb     = (const float*)d_in[8];
  const float* fuseg  = (const float*)d_in[9];
  const float* fuseb  = (const float*)d_in[10];
  const float* normg  = (const float*)d_in[11];
  const float* inw    = (const float*)d_in[12];
  const float* convw  = (const float*)d_in[13];
  const float* convb  = (const float*)d_in[14];
  const float* dtb    = (const float*)d_in[15];
  const float* alog   = (const float*)d_in[16];
  const float* Dpar   = (const float*)d_in[17];
  const float* outw   = (const float*)d_in[18];
  const float* manw   = (const float*)d_in[19];
  const float* manb   = (const float*)d_in[20];
  const float* fing   = (const float*)d_in[21];
  const float* finb   = (const float*)d_in[22];
  const float* onw1   = (const float*)d_in[23];
  const float* onb1   = (const float*)d_in[24];
  const float* onw2   = (const float*)d_in[25];
  const float* onb2   = (const float*)d_in[26];
  const float* offw1  = (const float*)d_in[27];
  const float* offb1  = (const float*)d_in[28];
  const float* offw2  = (const float*)d_in[29];
  const float* offb2  = (const float*)d_in[30];
  float* out = (float*)d_out;

  const size_t T = TOKENSC;
  float* ws = (float*)d_ws;
  size_t o = 0;
  float* EPI = ws + o; o += T*512;            // epi_f, later fwd-out
  float* G1  = ws + o; o += T*512;            // gate,  later bwd-out
  float* G2  = ws + o; o += T*512;            // featp, later xm
  float* X   = ws + o; o += T*512;
  float* XN  = ws + o; o += T*512;
  float* XR  = ws + o; o += T*512;            // flipped xn / manifold tmp
  float* ZX  = ws + o; o += T*(size_t)D_IN_PROJC;
  float* XBC = ws + o; o += T*(size_t)CONV_DIMC;
  float* DTX = ws + o; o += T*(size_t)D_INNERC;
  float* Y   = ws + o; o += T*(size_t)D_INNERC;
  float* DT  = ws + o; o += T*NHEADSC;
  float* SEG = ws + o; o += T*NHEADSC;
  float* CBB = ws + o; o += (size_t)BATCHC*NCHUNKC*CHUNKC*CHUNKC;
  float* ST  = ws + o; o += (size_t)BATCHC*NCHUNKC*NHEADSC*HEADDIMC*D_STATEC;
  float* LAT = ws + o; o += BATCHC*512;
  float* H1  = ws + o; o += BATCHC*256 + 8;
  u16*   ABF = (u16*)(ws + o); o += T*512;                       // T*1024 u16
  u16*   BBF = (u16*)(ws + o); o += (size_t)(512*2320/2 + 16);   // max weight, transposed
  float* FWD = EPI; float* BWD = G1; float* XM = G2; float* MT = XR;

  auto gg = [](size_t M, int N){ return dim3((unsigned)((N + 63)/64), (unsigned)((M + 63)/64), 1); };
  auto packA = [&](const float* src, long n){
    k_pack<<<(int)((n + 255)/256), 256, 0, stream>>>(src, ABF, n);
  };
  auto packB = [&](const float* wsrc, int K, int N){
    k_packT<<<dim3((N + 31)/32, (K + 31)/32), 256, 0, stream>>>(wsrc, BBF, K, N);
  };

  // ---- input fusion stage ----
  k_embed<<<(int)T, 256, 0, stream>>>(dna, emb, X);
  k_epiconv<<<(int)T, 256, 0, stream>>>(epi, fcw, fcb, EPI);
  packA(X, T*512); packB(gw, 512, 512);
  gemm_bf16<1><<<gg(T,512), 256, 0, stream>>>(ABF, BBF, gb, G1, 512, (int)T, 512, 512);
  packA(EPI, T*512); packB(fw, 512, 512);
  gemm_bf16<0><<<gg(T,512), 256, 0, stream>>>(ABF, BBF, fb, G2, 512, (int)T, 512, 512);
  k_fuse_ln<<<(int)T, 256, 0, stream>>>(X, G1, G2, fuseg, fuseb);

  // ---- mamba layers ----
  for (int l = 0; l < N_LAYERSC; ++l){
    k_rms<<<(int)T, 256, 0, stream>>>(X, normg + (size_t)l*512, XN);
    for (int dir = 0; dir < 2; ++dir){
      int ld = l*2 + dir;
      const float* xin  = XN;
      float*       ydir = FWD;
      if (dir == 1){
        k_flip<<<(int)T, 256, 0, stream>>>(XN, XR);
        xin = XR; ydir = BWD;
      }
      // in_proj
      packA(xin, T*512); packB(inw + (size_t)ld*512*D_IN_PROJC, 512, D_IN_PROJC);
      gemm_bf16<0><<<gg(T, D_IN_PROJC), 256, 0, stream>>>(ABF, BBF, nullptr, ZX, D_IN_PROJC,
                                                          (int)T, D_IN_PROJC, 512);
      // depthwise conv + silu, dt softplus + cumulative decay
      k_conv <<<(int)T, 256, 0, stream>>>(ZX, convw + (size_t)ld*CONV_DIMC*DKWC, convb + (size_t)ld*CONV_DIMC, XBC);
      k_dtseg<<<BATCHC*NCHUNKC, 16, 0, stream>>>(ZX, dtb + ld*NHEADSC, alog + ld*NHEADSC, DT, SEG);
      k_dtx  <<<(int)T, 256, 0, stream>>>(DT, XBC, DTX);
      // SSD
      k_cb    <<<dim3(32, BATCHC*NCHUNKC), 256, 0, stream>>>(XBC, CBB);
      k_yintra<<<dim3(8, BATCHC*NCHUNKC, NHEADSC), 256, 0, stream>>>(CBB, SEG, DTX, Y);
      k_states<<<dim3(4, BATCHC*NCHUNKC, NHEADSC), 256, 0, stream>>>(SEG, DTX, XBC, ST);
      k_scan  <<<BATCHC*NHEADSC, 256, 0, stream>>>(SEG, ST);
      k_yinter<<<dim3(8, BATCHC*NCHUNKC, NHEADSC), 256, 0, stream>>>(SEG, XBC, ST, Dpar + ld*NHEADSC, Y);
      // gating + out_proj
      k_gatemul<<<(int)T, 256, 0, stream>>>(ZX, Y);
      packA(Y, T*1024); packB(outw + (size_t)ld*D_INNERC*512, D_INNERC, 512);
      gemm_bf16<0><<<gg(T,512), 256, 0, stream>>>(ABF, BBF, nullptr, ydir, 512, (int)T, 512, D_INNERC);
    }
    k_combine<<<(int)T, 256, 0, stream>>>(X, FWD, BWD, XM);
    packA(XM, T*512); packB(manw + (size_t)l*512*512, 512, 512);
    gemm_bf16<0><<<gg(T,512), 256, 0, stream>>>(ABF, BBF, manb + (size_t)l*512, MT, 512, (int)T, 512, 512);
    k_tanhres<<<(int)T, 256, 0, stream>>>(XM, MT, X);
  }

  // ---- head ----
  k_ln_out<<<(int)T, 256, 0, stream>>>(X, fing, finb, XN);
  k_pool<<<BATCHC*512, 256, 0, stream>>>(XN, LAT);
  packA(LAT, BATCHC*512); packB(onw1, 512, 256);
  gemm_bf16<2><<<gg(BATCHC,256), 256, 0, stream>>>(ABF, BBF, onb1, H1, 256, BATCHC, 256, 512);
  k_head2<<<BATCHC, 256, 0, stream>>>(H1, onw2, onb2, out + 0);
  packB(offw1, 512, 256);
  gemm_bf16<2><<<gg(BATCHC,256), 256, 0, stream>>>(ABF, BBF, offb1, H1, 256, BATCHC, 256, 512);
  k_head2<<<BATCHC, 256, 0, stream>>>(H1, offw2, offb2, out + 2);
  k_copy<<<4, 256, 0, stream>>>(LAT, out + 4, BATCHC*512);
}
